// GCN_38242388804050
// MI455X (gfx1250) — matrix operand order repaired and checked by reading, not compiled, not run on hardware
//
#include <hip/hip_runtime.h>
#include <hip/hip_bf16.h>

// ---------------------------------------------------------------------------
// 2-layer GCN forward for MI455X (gfx1250, wave32).
//   h1 = relu( D^-1/2 A D^-1/2 (X W1) + b1 )
//   out = log_softmax( D^-1/2 A D^-1/2 (h1 W2) + b2 )
// GEMM1 runs on v_wmma_f32_16x16x32_bf16 with a bf16 hi/lo split (3 WMMAs)
// for ~fp32 accuracy. W1 fragments are pre-packed once into the per-lane WMMA
// register layout so the GEMM hot loop is pure b128 loads + WMMA.
// Edge phases are gather + global f32 atomic scatter-add (roofline-bound).
// ---------------------------------------------------------------------------

#define N_FEAT   128
#define N_HID    50
#define N_CLS    2

typedef __attribute__((ext_vector_type(16))) __bf16 v16bf;
typedef __attribute__((ext_vector_type(8)))  float  v8f;

__device__ __forceinline__ unsigned short f32_to_bf16_rne(float f) {
    union { float f; unsigned int u; } v; v.f = f;
    unsigned int u = v.u;
    u += 0x7FFFu + ((u >> 16) & 1u);     // round-to-nearest-even
    return (unsigned short)(u >> 16);
}
__device__ __forceinline__ float bf16_bits_to_f32(unsigned short h) {
    union { unsigned int u; float f; } v; v.u = ((unsigned int)h) << 16;
    return v.f;
}
__device__ __forceinline__ __bf16 bits_to_bf16(unsigned short h) {
    union { unsigned short u; __bf16 b; } v; v.u = h; return v.b;
}

// ---------------------------------------------------------------------------
// 0) init: acc buffers = 0, deg = 1 (self-loop pre-counted)
// ---------------------------------------------------------------------------
__global__ void gcn_init_kernel(float* __restrict__ deg,
                                float* __restrict__ acc1,
                                float* __restrict__ acc2,
                                int n, long long na1, int na2) {
    long long t = (long long)blockIdx.x * blockDim.x + threadIdx.x;
    long long stride = (long long)gridDim.x * blockDim.x;
    for (long long i = t; i < na1; i += stride) acc1[i] = 0.0f;
    for (long long i = t; i < na2; i += stride) acc2[i] = 0.0f;
    for (long long i = t; i < n;   i += stride) deg[i]  = 1.0f;
}

// ---------------------------------------------------------------------------
// 1) degree: deg[dst] += 1 per edge
// ---------------------------------------------------------------------------
__global__ void gcn_degree_kernel(const int* __restrict__ dst,
                                  float* __restrict__ deg, int nE) {
    int t = blockIdx.x * blockDim.x + threadIdx.x;
    int stride = gridDim.x * blockDim.x;
    for (int e = t; e < nE; e += stride) atomicAdd(&deg[dst[e]], 1.0f);
}

// 2) dinv = rsqrt(deg)  (deg >= 1 always: self-loops)
__global__ void gcn_dinv_kernel(float* __restrict__ deg, int n) {
    int t = blockIdx.x * blockDim.x + threadIdx.x;
    if (t < n) deg[t] = __frsqrt_rn(deg[t]);
}

// ---------------------------------------------------------------------------
// 2b) Pre-pack W1 into B-fragment layout (bf16 hi/lo), once per call.
//     Fragment (ks, nt): lane -> N = nt*16 + (lane&15);
//     element j (VGPR j/2) -> K = ks*32 + (lane>>4)*16 + j.
//     Storage: [frag(0..15)][lane(0..31)][j(0..15)] ushort, hi and lo planes.
// ---------------------------------------------------------------------------
__global__ void gcn_prepw_kernel(const float* __restrict__ W1,
                                 unsigned short* __restrict__ whi,
                                 unsigned short* __restrict__ wlo) {
    const int t = blockIdx.x * blockDim.x + threadIdx.x;  // 512 = 16 frags * 32 lanes
    if (t >= 16 * 32) return;
    const int lane = t & 31;
    const int frag = t >> 5;
    const int ks = frag >> 2;
    const int nt = frag & 3;
    const int n    = nt * 16 + (lane & 15);
    const int half = lane >> 4;
    #pragma unroll
    for (int j = 0; j < 16; ++j) {
        const int k = ks * 32 + half * 16 + j;
        float f = (n < N_HID) ? W1[k * N_HID + n] : 0.0f;
        unsigned short h = f32_to_bf16_rne(f);
        float rem = f - bf16_bits_to_f32(h);
        whi[t * 16 + j] = h;
        wlo[t * 16 + j] = f32_to_bf16_rne(rem);
    }
}

// ---------------------------------------------------------------------------
// 3) GEMM1 via WMMA:  hs1[i,:] = (X @ W1)[i,:] * dinv[i]
//    One wave computes one 16-row M-tile across N=64 (4 tiles, cols>=50 dropped).
//    K = 128 -> 4 steps of 32. bf16 hi/lo split: 3 WMMAs per (kstep, ntile).
//    B fragments are loaded pre-packed (2x b128 per fragment, L2-resident).
// ---------------------------------------------------------------------------
__global__ void gcn_gemm1_wmma_kernel(const float* __restrict__ x,
                                      const unsigned short* __restrict__ whi,
                                      const unsigned short* __restrict__ wlo,
                                      const float* __restrict__ dinv,
                                      float* __restrict__ hs1,
                                      int nNodes) {
    const int lane  = threadIdx.x & 31;
    const int wave  = threadIdx.x >> 5;
    const int tiles = (nNodes + 15) >> 4;
    const int tile  = blockIdx.x * (blockDim.x >> 5) + wave;
    if (tile >= tiles) return;                 // wave-uniform: EXEC stays all-1s

    const int m0   = tile << 4;
    const int half = lane >> 4;                // 0: lanes 0-15, 1: lanes 16-31
    int row = m0 + (lane & 15);
    if (row >= nNodes) row = nNodes - 1;       // clamp loads; stores guarded

    const v16bf* whiF = (const v16bf*)whi;     // 32 B per (frag, lane)
    const v16bf* wloF = (const v16bf*)wlo;

    v8f acc[4];
    #pragma unroll
    for (int nt = 0; nt < 4; ++nt) acc[nt] = v8f{0,0,0,0,0,0,0,0};

    #pragma unroll
    for (int ks = 0; ks < 4; ++ks) {
        // ---- A fragment (16x32 bf16), contiguous k order (same as packed B):
        //      element e lives in VGPR e/2; K = half*16 + e
        v16bf ahi, alo;
        const float* pa = x + (long long)row * N_FEAT + ks * 32;
        #pragma unroll
        for (int g = 0; g < 2; ++g) {          // two contiguous 8-float runs
            const int kb = half * 16 + g * 8;  // 32-B aligned
            const float4 f0 = *(const float4*)(pa + kb);
            const float4 f1 = *(const float4*)(pa + kb + 4);
            const float fv[8] = {f0.x, f0.y, f0.z, f0.w, f1.x, f1.y, f1.z, f1.w};
            #pragma unroll
            for (int j = 0; j < 8; ++j) {
                unsigned short h = f32_to_bf16_rne(fv[j]);
                float rem = fv[j] - bf16_bits_to_f32(h);
                ahi[g * 8 + j] = bits_to_bf16(h);
                alo[g * 8 + j] = bits_to_bf16(f32_to_bf16_rne(rem));
            }
        }
        #pragma unroll
        for (int nt = 0; nt < 4; ++nt) {
            const int fidx = (ks * 4 + nt) * 32 + lane;
            const v16bf bhi = whiF[fidx];
            const v16bf blo = wloF[fidx];
            acc[nt] = __builtin_amdgcn_wmma_f32_16x16x32_bf16(
                          false, ahi, false, bhi, (short)0, acc[nt], false, false);
            acc[nt] = __builtin_amdgcn_wmma_f32_16x16x32_bf16(
                          false, ahi, false, blo, (short)0, acc[nt], false, false);
            acc[nt] = __builtin_amdgcn_wmma_f32_16x16x32_bf16(
                          false, alo, false, bhi, (short)0, acc[nt], false, false);
        }
    }

    // ---- store C (16x16 f32 layout: VGPR r -> row r + half*8, col = lane&15),
    //      fused with dinv row scaling: hs1 = (XW1) * dinv[row]
    #pragma unroll
    for (int nt = 0; nt < 4; ++nt) {
        const int col = nt * 16 + (lane & 15);
        if (col < N_HID) {
            #pragma unroll
            for (int r = 0; r < 8; ++r) {
                const int m = m0 + r + half * 8;
                if (m < nNodes)
                    hs1[(long long)m * N_HID + col] = acc[nt][r] * dinv[m];
            }
        }
    }
}

// ---------------------------------------------------------------------------
// 4) Edge phase L1: wave per edge, acc1[dst,:] += hs1[src,:]  (50 floats)
// ---------------------------------------------------------------------------
__global__ void gcn_edge_agg1_kernel(const int* __restrict__ src,
                                     const int* __restrict__ dst,
                                     const float* __restrict__ hs1,
                                     float* __restrict__ acc1, int nE) {
    const int lane   = threadIdx.x & 31;
    const int wave   = (blockIdx.x * blockDim.x + threadIdx.x) >> 5;
    const int nWaves = (gridDim.x * blockDim.x) >> 5;
    for (int e = wave; e < nE; e += nWaves) {
        const int s = src[e];
        const int d = dst[e];
        const float* hp = hs1 + (long long)s * N_HID;
        float* ap       = acc1 + (long long)d * N_HID;
        atomicAdd(ap + lane, hp[lane]);
        if (lane < N_HID - 32)
            atomicAdd(ap + lane + 32, hp[lane + 32]);
    }
}

// ---------------------------------------------------------------------------
// 5) Node epilogue L1 + fused GEMM2: wave per node.
//    h = relu(dinv*(acc1 + hs1_selfloop) + b1);  hs2 = (h @ W2) * dinv
// ---------------------------------------------------------------------------
__global__ void gcn_node1_kernel(const float* __restrict__ acc1,
                                 const float* __restrict__ hs1,
                                 const float* __restrict__ dinv,
                                 const float* __restrict__ b1,
                                 const float* __restrict__ W2,
                                 float* __restrict__ hs2, int nNodes) {
    const int lane   = threadIdx.x & 31;
    const int wave   = (blockIdx.x * blockDim.x + threadIdx.x) >> 5;
    const int nWaves = (gridDim.x * blockDim.x) >> 5;
    for (int v = wave; v < nNodes; v += nWaves) {
        const float di = dinv[v];
        const float* ap = acc1 + (long long)v * N_HID;
        const float* hp = hs1 + (long long)v * N_HID;
        float p0 = 0.0f, p1 = 0.0f;
        {
            float h = di * (ap[lane] + hp[lane]) + b1[lane];
            h = fmaxf(h, 0.0f);
            p0 += h * W2[lane * N_CLS + 0];
            p1 += h * W2[lane * N_CLS + 1];
        }
        if (lane < N_HID - 32) {
            const int f = lane + 32;
            float h = di * (ap[f] + hp[f]) + b1[f];
            h = fmaxf(h, 0.0f);
            p0 += h * W2[f * N_CLS + 0];
            p1 += h * W2[f * N_CLS + 1];
        }
        #pragma unroll
        for (int off = 16; off > 0; off >>= 1) {
            p0 += __shfl_xor(p0, off, 32);
            p1 += __shfl_xor(p1, off, 32);
        }
        if (lane == 0) {
            hs2[v * N_CLS + 0] = p0 * di;
            hs2[v * N_CLS + 1] = p1 * di;
        }
    }
}

// 6) Edge phase L2: thread per edge (2 floats)
__global__ void gcn_edge_agg2_kernel(const int* __restrict__ src,
                                     const int* __restrict__ dst,
                                     const float* __restrict__ hs2,
                                     float* __restrict__ acc2, int nE) {
    int t = blockIdx.x * blockDim.x + threadIdx.x;
    int stride = gridDim.x * blockDim.x;
    for (int e = t; e < nE; e += stride) {
        const int s = src[e];
        const int d = dst[e];
        atomicAdd(&acc2[d * N_CLS + 0], hs2[s * N_CLS + 0]);
        atomicAdd(&acc2[d * N_CLS + 1], hs2[s * N_CLS + 1]);
    }
}

// 7) Node epilogue L2 + log_softmax over 2 classes
__global__ void gcn_node2_kernel(const float* __restrict__ acc2,
                                 const float* __restrict__ hs2,
                                 const float* __restrict__ dinv,
                                 const float* __restrict__ b2,
                                 float* __restrict__ out, int nNodes) {
    int t = blockIdx.x * blockDim.x + threadIdx.x;
    int stride = gridDim.x * blockDim.x;
    for (int v = t; v < nNodes; v += stride) {
        const float di = dinv[v];
        float o0 = di * (acc2[v * N_CLS + 0] + hs2[v * N_CLS + 0]) + b2[0];
        float o1 = di * (acc2[v * N_CLS + 1] + hs2[v * N_CLS + 1]) + b2[1];
        const float m = fmaxf(o0, o1);
        const float lse = m + logf(expf(o0 - m) + expf(o1 - m));
        out[v * N_CLS + 0] = o0 - lse;
        out[v * N_CLS + 1] = o1 - lse;
    }
}

// ---------------------------------------------------------------------------
extern "C" void kernel_launch(void* const* d_in, const int* in_sizes, int n_in,
                              void* d_out, int out_size, void* d_ws, size_t ws_size,
                              hipStream_t stream) {
    const float* x  = (const float*)d_in[0];
    const int*   ei = (const int*)d_in[1];
    const float* W1 = (const float*)d_in[2];
    const float* b1 = (const float*)d_in[3];
    const float* W2 = (const float*)d_in[4];
    const float* b2 = (const float*)d_in[5];
    float* out = (float*)d_out;

    const int nNodes = in_sizes[0] / N_FEAT;     // 100000
    const int nEdges = in_sizes[1] / 2;          // 3200000
    const int* src = ei;
    const int* dst = ei + nEdges;

    // workspace layout (f32 units), 256-B aligned regions
    float* ws = (float*)d_ws;
    size_t off = 0;
    auto take = [&](size_t n) { float* p = ws + off; off += (n + 63) & ~(size_t)63; return p; };
    float* deg  = take((size_t)nNodes);                 // becomes dinv in-place
    float* hs1  = take((size_t)nNodes * N_HID);
    float* acc1 = take((size_t)nNodes * N_HID);
    float* hs2  = take((size_t)nNodes * N_CLS);
    float* acc2 = take((size_t)nNodes * N_CLS);
    unsigned short* whi = (unsigned short*)take(16 * 32 * 16 / 2);  // 16 KB
    unsigned short* wlo = (unsigned short*)take(16 * 32 * 16 / 2);  // 16 KB
    (void)ws_size; (void)n_in; (void)out_size;

    const long long nAcc1 = (long long)nNodes * N_HID;
    const int       nAcc2 = nNodes * N_CLS;

    // 0) init workspace (every call: accumulators must start at zero)
    gcn_init_kernel<<<2048, 256, 0, stream>>>(deg, acc1, acc2, nNodes, nAcc1, nAcc2);

    // 1) degree (self-loop pre-counted via deg=1)
    gcn_degree_kernel<<<2048, 256, 0, stream>>>(dst, deg, nEdges);

    // 2) dinv = rsqrt(deg)
    gcn_dinv_kernel<<<(nNodes + 255) / 256, 256, 0, stream>>>(deg, nNodes);

    // 2b) pre-pack W1 bf16 hi/lo fragments (two blocks)
    gcn_prepw_kernel<<<2, 256, 0, stream>>>(W1, whi, wlo);

    // 3) GEMM1 on WMMA (one 16-row tile per wave; 4 waves per block)
    {
        const int tiles  = (nNodes + 15) / 16;
        const int blocks = (tiles + 3) / 4;
        gcn_gemm1_wmma_kernel<<<blocks, 128, 0, stream>>>(x, whi, wlo, deg, hs1, nNodes);
    }

    // 4) edge gather + atomic scatter-add, layer 1 (wave per edge)
    gcn_edge_agg1_kernel<<<8192, 256, 0, stream>>>(src, dst, hs1, acc1, nEdges);

    // 5) node epilogue L1 + fused GEMM2 (wave per node)
    {
        const int blocks = (nNodes + 7) / 8;   // 8 waves per 256-thread block
        gcn_node1_kernel<<<blocks, 256, 0, stream>>>(acc1, hs1, deg, b1, W2, hs2, nNodes);
    }

    // 6) edge phase, layer 2 (thread per edge)
    gcn_edge_agg2_kernel<<<4096, 256, 0, stream>>>(src, dst, hs2, acc2, nEdges);

    // 7) log-softmax output
    gcn_node2_kernel<<<(nNodes + 255) / 256, 256, 0, stream>>>(acc2, hs2, deg, b2, out, nNodes);
}
